// SimpleGCN_29557964931482
// MI455X (gfx1250) — matrix operand order repaired and checked by reading, not compiled, not run on hardware
//
#include <hip/hip_runtime.h>
#include <stdint.h>

typedef __attribute__((ext_vector_type(16))) __bf16 v16bf;
typedef __attribute__((ext_vector_type(8)))  float  v8f;

#define FEAT 128

union BF16x16 { v16bf v; uint4 q[2]; };

static __device__ __forceinline__ unsigned short f2bf(float f) {
  // round-to-nearest-even f32 -> bf16
  unsigned int u = __float_as_uint(f);
  unsigned int r = u + 0x7FFFu + ((u >> 16) & 1u);
  return (unsigned short)(r >> 16);
}

static __device__ __forceinline__ unsigned int pack2bf(float lo, float hi) {
  return (unsigned int)f2bf(lo) | ((unsigned int)f2bf(hi) << 16);
}

// ---------------- degree / normalization ----------------
__global__ void k_deg_init(float* deg, int N) {
  int i = blockIdx.x * blockDim.x + threadIdx.x;
  if (i < N) deg[i] = 1.0f;  // self-loop
}

__global__ void k_deg_accum(const long long* __restrict__ dst, float* deg, int E) {
  int e = blockIdx.x * blockDim.x + threadIdx.x;
  if (e < E) atomicAdd(&deg[(int)dst[e]], 1.0f);
}

__global__ void k_deg_rsqrt(float* deg, int N) {
  int i = blockIdx.x * blockDim.x + threadIdx.x;
  if (i < N) deg[i] = rsqrtf(deg[i]);  // deg >= 1 always
}

// ---------------- weight pre-conversion: Wbt[n*128+k] = bf16(W[k*128+n]) -------------
__global__ void k_w2bf(const float* __restrict__ W, unsigned short* __restrict__ Wbt) {
  int idx = blockIdx.x * blockDim.x + threadIdx.x;
  if (idx >= FEAT * FEAT) return;
  int n = idx >> 7, k = idx & (FEAT - 1);
  Wbt[idx] = f2bf(W[(size_t)k * FEAT + n]);
}

// ---------------- activation pre-conversion: f32 -> bf16 row-major -------------------
__global__ void k_a2bf(const float* __restrict__ A, unsigned short* __restrict__ Ab, int N) {
  int idx = blockIdx.x * blockDim.x + threadIdx.x;   // N*32 float4 chunks
  if (idx >= N * 32) return;
  float4 v = ((const float4*)A)[idx];
  uint2 o;
  o.x = pack2bf(v.x, v.y);
  o.y = pack2bf(v.z, v.w);
  ((uint2*)Ab)[idx] = o;
}

// ---------------- dense GEMM: C[N,128] = A[N,128] @ W[128,128] (bf16 WMMA, f32 acc) ---
// A fragment: lane holds row (lane&15), K = {ab..ab+7, ab+16..ab+23}, ab = k0 + half*8
// B fragment: lane holds col n0+(lane&15), K = {kb..kb+7, kb+16..kb+23}, kb = k0 + half*8
__global__ __launch_bounds__(256) void k_gemm_bf16(const unsigned short* __restrict__ Ab,
                                                   const unsigned short* __restrict__ Wbt,
                                                   float* __restrict__ C, int N) {
  const int wave = threadIdx.x >> 5;   // 0..7 -> 16-col tile
  const int lane = threadIdx.x & 31;
  const int half = lane >> 4;          // 0 | 1
  const int l15  = lane & 15;
  const int m0   = blockIdx.x * 16;
  const int n0   = wave * 16;

  int row = m0 + l15;
  if (row >= N) row = N - 1;           // clamp; out-of-range rows dropped at store
  const unsigned short* __restrict__ arow = Ab + (size_t)row * FEAT;
  const unsigned short* __restrict__ bcol = Wbt + (size_t)(n0 + l15) * FEAT;

  v8f acc = {};
  #pragma unroll
  for (int k0 = 0; k0 < FEAT; k0 += 32) {
    BF16x16 a, b;
    const int ab = k0 + half * 8;
    a.q[0] = *(const uint4*)(arow + ab);
    a.q[1] = *(const uint4*)(arow + ab + 16);
    const int kb = k0 + half * 8;
    b.q[0] = *(const uint4*)(bcol + kb);
    b.q[1] = *(const uint4*)(bcol + kb + 16);
    acc = __builtin_amdgcn_wmma_f32_16x16x32_bf16(false, a.v, false, b.v,
                                                  (short)0, acc, false, false);
  }
  // D layout: VGPR r -> row m0 + r + half*8, col n0 + (lane&15)
  #pragma unroll
  for (int r = 0; r < 8; ++r) {
    const int m = m0 + r + half * 8;
    if (m < N) C[(size_t)m * FEAT + n0 + l15] = acc[r];
  }
}

// ---------------- self-loop init: AGG = H * dinv^2 ----------------
__global__ void k_self_init(const float* __restrict__ H, const float* __restrict__ dinv,
                            float* __restrict__ AGG, int N) {
  int idx = blockIdx.x * blockDim.x + threadIdx.x;   // N*32 float4 chunks
  if (idx >= N * 32) return;
  int node = idx >> 5;
  float s = dinv[node];
  s *= s;
  float4 v = ((const float4*)H)[idx];
  v.x *= s; v.y *= s; v.z *= s; v.w *= s;
  ((float4*)AGG)[idx] = v;
}

// ---------------- edge scatter: one wave per edge, 4 floats/lane ----------------
__global__ __launch_bounds__(256) void k_edge_scatter(const long long* __restrict__ src,
                                                      const long long* __restrict__ dst,
                                                      const float* __restrict__ dinv,
                                                      const float* __restrict__ H,
                                                      float* __restrict__ AGG, int E) {
  int e = blockIdx.x * 8 + (threadIdx.x >> 5);
  if (e >= E) return;
  const int lane = threadIdx.x & 31;
  const int s = (int)src[e];
  const int d = (int)dst[e];
  const float c = dinv[s] * dinv[d];
  float4 v = ((const float4*)(H + (size_t)s * FEAT))[lane];
  float* o = AGG + (size_t)d * FEAT + lane * 4;
  atomicAdd(o + 0, v.x * c);
  atomicAdd(o + 1, v.y * c);
  atomicAdd(o + 2, v.z * c);
  atomicAdd(o + 3, v.w * c);
}

// ---------------- bias + relu (f32 out only) ----------------
__global__ void k_bias_relu(const float* __restrict__ AGG, const float* __restrict__ bias,
                            float* __restrict__ OUT, int N) {
  int idx = blockIdx.x * blockDim.x + threadIdx.x;   // N*32 float4 chunks
  if (idx >= N * 32) return;
  float4 v = ((const float4*)AGG)[idx];
  float4 b = ((const float4*)bias)[idx & 31];
  v.x = fmaxf(v.x + b.x, 0.0f);
  v.y = fmaxf(v.y + b.y, 0.0f);
  v.z = fmaxf(v.z + b.z, 0.0f);
  v.w = fmaxf(v.w + b.w, 0.0f);
  ((float4*)OUT)[idx] = v;
}

// ---------------- bias + relu, dual output (f32 for edge gather, bf16 for next GEMM) --
__global__ void k_bias_relu_bf(const float* __restrict__ AGG, const float* __restrict__ bias,
                               float* __restrict__ OUT, unsigned short* __restrict__ OUTb,
                               int N) {
  int idx = blockIdx.x * blockDim.x + threadIdx.x;   // N*32 float4 chunks
  if (idx >= N * 32) return;
  float4 v = ((const float4*)AGG)[idx];
  float4 b = ((const float4*)bias)[idx & 31];
  v.x = fmaxf(v.x + b.x, 0.0f);
  v.y = fmaxf(v.y + b.y, 0.0f);
  v.z = fmaxf(v.z + b.z, 0.0f);
  v.w = fmaxf(v.w + b.w, 0.0f);
  ((float4*)OUT)[idx] = v;
  uint2 o;
  o.x = pack2bf(v.x, v.y);
  o.y = pack2bf(v.z, v.w);
  ((uint2*)OUTb)[idx] = o;
}

// ---------------- pooling ----------------
__global__ void k_pool_init(float* sums, float* cnt, int G) {
  int i = blockIdx.x * blockDim.x + threadIdx.x;
  int tot = G * FEAT;
  if (i < tot) sums[i] = 0.0f;
  else if (i < tot + G) cnt[i - tot] = 0.0f;
}

__global__ __launch_bounds__(256) void k_pool_accum(const float* __restrict__ H,
                                                    const long long* __restrict__ batch,
                                                    float* __restrict__ sums,
                                                    float* __restrict__ cnt, int N) {
  int n = blockIdx.x * 8 + (threadIdx.x >> 5);
  if (n >= N) return;
  const int lane = threadIdx.x & 31;
  const int g = (int)batch[n];
  float4 v = ((const float4*)(H + (size_t)n * FEAT))[lane];
  float* s = sums + (size_t)g * FEAT + lane * 4;
  atomicAdd(s + 0, v.x);
  atomicAdd(s + 1, v.y);
  atomicAdd(s + 2, v.z);
  atomicAdd(s + 3, v.w);
  if (lane == 0) atomicAdd(&cnt[g], 1.0f);
}

__global__ void k_pool_out(const float* __restrict__ sums, const float* __restrict__ cnt,
                           const float* __restrict__ Wout, const float* __restrict__ bout,
                           float* __restrict__ out) {
  __shared__ float red[FEAT];
  const int g = blockIdx.x;
  const int f = threadIdx.x;
  red[f] = sums[(size_t)g * FEAT + f] * Wout[f];
  __syncthreads();
  for (int s = FEAT / 2; s > 0; s >>= 1) {
    if (f < s) red[f] += red[f + s];
    __syncthreads();
  }
  if (f == 0) out[g] = red[0] / fmaxf(cnt[g], 1.0f) + bout[0];
}

// ---------------- host launch ----------------
extern "C" void kernel_launch(void* const* d_in, const int* in_sizes, int n_in,
                              void* d_out, int out_size, void* d_ws, size_t ws_size,
                              hipStream_t stream) {
  const float*     x     = (const float*)d_in[0];
  const long long* edge  = (const long long*)d_in[1];
  const long long* batch = (const long long*)d_in[2];
  const float*     W1    = (const float*)d_in[3];
  const float*     b1    = (const float*)d_in[4];
  const float*     W2    = (const float*)d_in[5];
  const float*     b2    = (const float*)d_in[6];
  const float*     Wout  = (const float*)d_in[7];
  const float*     bout  = (const float*)d_in[8];
  float*           out   = (float*)d_out;

  const int N = in_sizes[0] / FEAT;
  const int E = in_sizes[1] / 2;
  const int G = out_size;

  const long long* src = edge;       // edge_index[0]
  const long long* dst = edge + E;   // edge_index[1]

  // workspace carve-out (256B aligned regions)
  char* base = (char*)d_ws;
  size_t off = 0;
  auto carve = [&](size_t bytes) -> void* {
    void* p = base + off;
    off = (off + bytes + 255) & ~(size_t)255;
    return p;
  };
  float*          dinv = (float*)carve((size_t)N * sizeof(float));
  float*          bufA = (float*)carve((size_t)N * FEAT * sizeof(float));
  float*          bufB = (float*)carve((size_t)N * FEAT * sizeof(float));
  unsigned short* Ab   = (unsigned short*)carve((size_t)N * FEAT * sizeof(unsigned short));
  unsigned short* Wb1  = (unsigned short*)carve((size_t)FEAT * FEAT * sizeof(unsigned short));
  unsigned short* Wb2  = (unsigned short*)carve((size_t)FEAT * FEAT * sizeof(unsigned short));
  float*          sums = (float*)carve((size_t)G * FEAT * sizeof(float));
  float*          cnt  = (float*)carve((size_t)G * sizeof(float));
  (void)ws_size; (void)n_in;

  const int TB = 256;
  const int gN    = (N + TB - 1) / TB;
  const int gE    = (E + TB - 1) / TB;
  const int gNF4  = (N * 32 + TB - 1) / TB;        // N*128 floats as float4
  const int gGemm = (N + 15) / 16;
  const int gEw   = (E + 7) / 8;                   // wave-per-edge
  const int gNw   = (N + 7) / 8;                   // wave-per-node
  const int gW    = (FEAT * FEAT + TB - 1) / TB;

  // normalization coefficients
  k_deg_init  <<<gN, TB, 0, stream>>>(dinv, N);
  k_deg_accum <<<gE, TB, 0, stream>>>(dst, dinv, E);
  k_deg_rsqrt <<<gN, TB, 0, stream>>>(dinv, N);

  // pre-convert weights (transposed bf16) and input activations (bf16)
  k_w2bf <<<gW,   TB, 0, stream>>>(W1, Wb1);
  k_w2bf <<<gW,   TB, 0, stream>>>(W2, Wb2);
  k_a2bf <<<gNF4, TB, 0, stream>>>(x, Ab, N);

  // layer 1: relu( scatter(norm * (x@W1)) + b1 )
  k_gemm_bf16   <<<gGemm, TB, 0, stream>>>(Ab, Wb1, bufA, N);
  k_self_init   <<<gNF4,  TB, 0, stream>>>(bufA, dinv, bufB, N);
  k_edge_scatter<<<gEw,   TB, 0, stream>>>(src, dst, dinv, bufA, bufB, E);
  k_bias_relu_bf<<<gNF4,  TB, 0, stream>>>(bufB, b1, bufA, Ab, N);

  // layer 2
  k_gemm_bf16   <<<gGemm, TB, 0, stream>>>(Ab, Wb2, bufB, N);
  k_self_init   <<<gNF4,  TB, 0, stream>>>(bufB, dinv, bufA, N);
  k_edge_scatter<<<gEw,   TB, 0, stream>>>(src, dst, dinv, bufB, bufA, E);
  k_bias_relu   <<<gNF4,  TB, 0, stream>>>(bufA, b2, bufB, N);

  // global mean pool + output projection
  k_pool_init <<<(G * (FEAT + 1) + TB - 1) / TB, TB, 0, stream>>>(sums, cnt, G);
  k_pool_accum<<<gNw, TB, 0, stream>>>(bufB, batch, sums, cnt, N);
  k_pool_out  <<<G, FEAT, 0, stream>>>(sums, cnt, Wout, bout, out);
}
